// T5EncoderSelfAttention_26396869001869
// MI455X (gfx1250) — matrix operand order repaired and checked by reading, not compiled, not run on hardware
//
#include <hip/hip_runtime.h>
#include <hip/hip_bf16.h>

#define NH   16
#define HD   64
#define EMB  1024
#define BSZ  4
#define SEQ  2048
#define MTOT (BSZ * SEQ)   // 8192

typedef __attribute__((ext_vector_type(16))) __bf16 v16bf;
typedef __attribute__((ext_vector_type(8)))  float  v8f;
typedef int v4i_n __attribute__((vector_size(16)));   // native vector int4

#define GLOBAL_AS __attribute__((address_space(1)))
#define LDS_AS    __attribute__((address_space(3)))

// ---- gfx1250 async global->LDS path (ASYNCcnt), with safe fallback --------
#if defined(__has_builtin)
#  if __has_builtin(__builtin_amdgcn_global_load_async_to_lds_b128)
#    define ASYNC_LDS 1
#  endif
#endif
#ifndef ASYNC_LDS
#  define ASYNC_LDS 0
#endif

#if ASYNC_LDS
#  if __has_builtin(__builtin_amdgcn_s_wait_asynccnt)
#    define WAIT_ASYNC() __builtin_amdgcn_s_wait_asynccnt(0)
#  else
#    define WAIT_ASYNC() asm volatile("s_wait_asynccnt 0" ::: "memory")
#  endif
#  define ASYNC_CP16(gsrc, ldst, OFF)                                         \
    __builtin_amdgcn_global_load_async_to_lds_b128(                           \
        (GLOBAL_AS v4i_n*)(gsrc), (LDS_AS v4i_n*)(ldst), (OFF), 0)
#else
#  define WAIT_ASYNC()
#endif

// ---- packed f32 -> bf16 conversion (v_cvt_pk_bf16_f32 when available) -----
#if defined(__has_builtin)
#  if __has_builtin(__builtin_amdgcn_cvt_pk_bf16_f32)
#    define HAVE_PK_BF16 1
#  endif
#endif
#ifndef HAVE_PK_BF16
#  define HAVE_PK_BF16 0
#endif

__device__ __forceinline__ unsigned short f32_to_bf16(float f) {
  unsigned int u = __builtin_bit_cast(unsigned int, f);
  u += 0x7FFFu + ((u >> 16) & 1u);          // round-to-nearest-even
  return (unsigned short)(u >> 16);
}

__device__ __forceinline__ unsigned int pack2_bf16(float a, float b) {
#if HAVE_PK_BF16
  typedef __attribute__((ext_vector_type(2))) __bf16 v2bf;
  v2bf r = __builtin_amdgcn_cvt_pk_bf16_f32(a, b);
  return __builtin_bit_cast(unsigned int, r);
#else
  return (unsigned int)f32_to_bf16(a) | ((unsigned int)f32_to_bf16(b) << 16);
#endif
}

union Frag32B {
  uint4 q[2];
  v16bf v;
};

// Build a 16x32 (A) or 32x16 (B) bf16 fragment from two contiguous 16B runs.
__device__ __forceinline__ v16bf ld_frag(const unsigned short* p0,
                                         const unsigned short* p1) {
  Frag32B f;
  f.q[0] = *(const uint4*)p0;
  f.q[1] = *(const uint4*)p1;
  return f.v;
}

__device__ __forceinline__ v8f wmma_bf16(v16bf a, v16bf b, v8f c) {
  return __builtin_amdgcn_wmma_f32_16x16x32_bf16(
      /*neg_a=*/false, a, /*neg_b=*/false, b,
      /*c_mod=*/(short)0, c, /*reuse_a=*/false, /*reuse_b=*/false);
}

// ---------------------------------------------------------------------------
// Kernel 1: fused QKV projection.  out = x @ W.T, stored bf16 as [b,h,s,d].
// Block = 256 threads (8 waves), tile 128(M) x 64(N), K-step 32.
// grid = (MTOT/128, EMB/64, 3)  z selects Wq/Wk/Wv.
// ---------------------------------------------------------------------------
__global__ __launch_bounds__(256)
void qkv_proj_kernel(const float* __restrict__ x,
                     const float* __restrict__ Wq,
                     const float* __restrict__ Wk,
                     const float* __restrict__ Wv,
                     unsigned short* __restrict__ q_ws,
                     unsigned short* __restrict__ k_ws,
                     unsigned short* __restrict__ v_ws) {
  __shared__ unsigned short xs[128][32];   // x tile, bf16
  __shared__ unsigned short wsh[64][32];   // W tile, bf16

  const int m0    = blockIdx.x * 128;
  const int n0    = blockIdx.y * 64;
  const int which = blockIdx.z;
  const float* W = (which == 0) ? Wq : ((which == 1) ? Wk : Wv);
  unsigned short* outp = (which == 0) ? q_ws : ((which == 1) ? k_ws : v_ws);

  const int tid  = threadIdx.x;
  const int wave = tid >> 5;
  const int lane = tid & 31;
  const int lr   = lane & 15;
  const int hg   = (lane >> 4) & 1;

  v8f acc[4];
#pragma unroll
  for (int i = 0; i < 4; ++i) acc[i] = (v8f)0.0f;

  for (int k0 = 0; k0 < EMB; k0 += 32) {
    // ---- stage x tile: 128x32 f32 -> bf16 (packed cvt) ----
    {
      const int r  = tid >> 1;
      const int c0 = (tid & 1) * 16;
      const float* src = x + (size_t)(m0 + r) * EMB + k0 + c0;
      alignas(16) unsigned int t32[8];
#pragma unroll
      for (int j = 0; j < 4; ++j) {
        float4 f = *(const float4*)(src + j * 4);
        t32[j * 2 + 0] = pack2_bf16(f.x, f.y);
        t32[j * 2 + 1] = pack2_bf16(f.z, f.w);
      }
      *(uint4*)&xs[r][c0]     = *(const uint4*)&t32[0];
      *(uint4*)&xs[r][c0 + 8] = *(const uint4*)&t32[4];
    }
    // ---- stage W tile: 64x32 f32 -> bf16 ----
    {
      const int r  = tid >> 2;
      const int c0 = (tid & 3) * 8;
      const float* src = W + (size_t)(n0 + r) * EMB + k0 + c0;
      alignas(16) unsigned int t32[4];
#pragma unroll
      for (int j = 0; j < 2; ++j) {
        float4 f = *(const float4*)(src + j * 4);
        t32[j * 2 + 0] = pack2_bf16(f.x, f.y);
        t32[j * 2 + 1] = pack2_bf16(f.z, f.w);
      }
      *(uint4*)&wsh[r][c0] = *(const uint4*)&t32[0];
    }
    __syncthreads();

    // A fragment: rows m0+wave*16 .. +15, K-chunk k0..k0+31 (ISA A-layout)
    const v16bf a = ld_frag(&xs[wave * 16 + lr][hg * 8],
                            &xs[wave * 16 + lr][16 + hg * 8]);
#pragma unroll
    for (int nt = 0; nt < 4; ++nt) {
      // B fragment: column n = nt*16+lr is W row n (out = x @ W.T)
      const v16bf b = ld_frag(&wsh[nt * 16 + lr][hg * 8],
                              &wsh[nt * 16 + lr][16 + hg * 8]);
      acc[nt] = wmma_bf16(a, b, acc[nt]);
    }
    __syncthreads();
  }

  // ---- scatter to [b,h,s,d] bf16; b and h are block constants ----
  {
    const int bb = m0 >> 11;            // / SEQ
    const int hh = n0 >> 6;             // / HD   (64-wide tile == one head)
    const int s0 = (m0 & (SEQ - 1)) + wave * 16 + hg * 8;
    unsigned short* obase =
        outp + (((size_t)bb * NH + hh) * SEQ + s0) * HD + lr;
#pragma unroll
    for (int nt = 0; nt < 4; ++nt)
#pragma unroll
      for (int i = 0; i < 8; ++i)
        obase[(size_t)i * HD + nt * 16] = f32_to_bf16(acc[nt][i]);
  }
}

// ---------------------------------------------------------------------------
// Kernel 2: flash attention with additive bias (unscaled scores, T5-style).
// Block = 256 threads (8 waves). Q tile 128 rows; key blocks of 32,
// double-buffered: async DMA of tile i+1 overlaps compute of tile i.
// grid = (SEQ/128, NH, BSZ)
// ---------------------------------------------------------------------------
__global__ __launch_bounds__(256)
void attn_kernel(const unsigned short* __restrict__ q_ws,
                 const unsigned short* __restrict__ k_ws,
                 const unsigned short* __restrict__ v_ws,
                 const float* __restrict__ bias,
                 unsigned short* __restrict__ attn_ws) {
  __shared__ unsigned short Qs[128][64];       // 16 KB
  __shared__ unsigned short Ks[2][32][64];     //  8 KB double-buffered
  __shared__ unsigned short Vts[2][64][32];    //  8 KB double-buffered [d][k]
  __shared__ unsigned short Ps[8][16][32];     //  8 KB wave-private P bounce

  const int q0 = blockIdx.x * 128;
  const int h  = blockIdx.y;
  const int b  = blockIdx.z;

  const int tid  = threadIdx.x;
  const int wave = tid >> 5;
  const int lane = tid & 31;
  const int lr   = lane & 15;
  const int hg   = (lane >> 4) & 1;

  const size_t headbase = ((size_t)b * NH + h) * SEQ * HD;
  const unsigned short* Qg = q_ws + headbase;
  const unsigned short* Kh = k_ws + headbase;
  const unsigned short* Vh = v_ws + headbase;

  // staging coordinates (same for every key block)
  const int sr  = tid >> 3;               // 0..31 key row
  const int sc0 = (tid & 7) * 8;          // d offset (16B granules)

  // ---- stage Q tile 128x64 bf16 (pure copy -> async DMA to LDS) ----
  {
    const int r  = tid >> 1;
    const int c0 = (tid & 1) * 32;
    const unsigned short* g = Qg + (size_t)(q0 + r) * HD + c0;
    unsigned short* l = &Qs[r][c0];
#if ASYNC_LDS
    ASYNC_CP16(g, l, 0);
    ASYNC_CP16(g, l, 16);
    ASYNC_CP16(g, l, 32);
    ASYNC_CP16(g, l, 48);
#else
    const uint4* src = (const uint4*)g;
    uint4* dst = (uint4*)l;
#pragma unroll
    for (int j = 0; j < 4; ++j) dst[j] = src[j];
#endif
  }

  // ---- preload key-block 0: K via async DMA, V into VGPRs ----
  uint4 vreg;
  {
#if ASYNC_LDS
    ASYNC_CP16(Kh + (size_t)sr * HD + sc0, &Ks[0][sr][sc0], 0);
#else
    *(uint4*)&Ks[0][sr][sc0] = *(const uint4*)(Kh + (size_t)sr * HD + sc0);
#endif
    vreg = *(const uint4*)(Vh + (size_t)sr * HD + sc0);
  }

  WAIT_ASYNC();      // Q tile (and K tile 0) resident
  __syncthreads();

  // Q fragments for both 32-wide d-chunks (held in registers for all keys)
  v16bf qf[2];
#pragma unroll
  for (int c = 0; c < 2; ++c)
    qf[c] = ld_frag(&Qs[wave * 16 + lr][c * 32 + hg * 8],
                    &Qs[wave * 16 + lr][c * 32 + 16 + hg * 8]);

  float mrow[8], lrow[8];
  v8f   oacc[4];
#pragma unroll
  for (int i = 0; i < 8; ++i) { mrow[i] = -3.0e38f; lrow[i] = 0.0f; }
#pragma unroll
  for (int i = 0; i < 4; ++i) oacc[i] = (v8f)0.0f;

  for (int kb = 0; kb < SEQ; kb += 32) {
    const int cur = (kb >> 5) & 1;

    WAIT_ASYNC();                        // K tile `cur` complete
    // transpose-store V for current tile from preloaded VGPRs
    {
      union { uint4 q; unsigned short s[8]; } u;
      u.q = vreg;
#pragma unroll
      for (int j = 0; j < 8; ++j) Vts[cur][sc0 + j][sr] = u.s[j];
    }
    __syncthreads();                     // staging of `cur` visible everywhere

    // ---- kick off tile cur^1 (overlaps with compute below) ----
    if (kb + 32 < SEQ) {
      const unsigned short* Kg = Kh + (size_t)(kb + 32) * HD;
      const unsigned short* Vg = Vh + (size_t)(kb + 32) * HD;
      __builtin_prefetch(Vg, 0, 1);      // global_prefetch_b8
#if ASYNC_LDS
      ASYNC_CP16(Kg + (size_t)sr * HD + sc0, &Ks[cur ^ 1][sr][sc0], 0);
#else
      *(uint4*)&Ks[cur ^ 1][sr][sc0] =
          *(const uint4*)(Kg + (size_t)sr * HD + sc0);
#endif
      vreg = *(const uint4*)(Vg + (size_t)sr * HD + sc0);
    }

    // ---- S = Q @ K^T  (two 16x16 key tiles, reduce over d in 2 chunks) ----
    v8f sacc[2];
    sacc[0] = (v8f)0.0f;
    sacc[1] = (v8f)0.0f;
#pragma unroll
    for (int kt = 0; kt < 2; ++kt) {
#pragma unroll
      for (int c = 0; c < 2; ++c) {
        const v16bf kf =
            ld_frag(&Ks[cur][kt * 16 + lr][c * 32 + hg * 8],
                    &Ks[cur][kt * 16 + lr][c * 32 + 16 + hg * 8]);
        sacc[kt] = wmma_bf16(qf[c], kf, sacc[kt]);
      }
    }

    // ---- additive relative-position bias (fp32, streamed once) ----
    const float* bs = bias +
        ((size_t)h * SEQ + (q0 + wave * 16 + hg * 8)) * SEQ + kb;
    __builtin_prefetch(bs + 32, 0, 0);
#pragma unroll
    for (int kt = 0; kt < 2; ++kt)
#pragma unroll
      for (int i = 0; i < 8; ++i)
        sacc[kt][i] += bs[(size_t)i * SEQ + kt * 16 + lr];

    // ---- online softmax update ----
#pragma unroll
    for (int i = 0; i < 8; ++i) {
      float v = fmaxf(sacc[0][i], sacc[1][i]);
#pragma unroll
      for (int moff = 8; moff >= 1; moff >>= 1)
        v = fmaxf(v, __shfl_xor(v, moff, 32));     // reduce over 16-lane group
      const float mnew = fmaxf(mrow[i], v);
      const float corr = __expf(mrow[i] - mnew);
      mrow[i] = mnew;
      const float p0 = __expf(sacc[0][i] - mnew);
      const float p1 = __expf(sacc[1][i] - mnew);
      lrow[i] = lrow[i] * corr + p0 + p1;
#pragma unroll
      for (int dt = 0; dt < 4; ++dt) oacc[dt][i] *= corr;
      sacc[0][i] = p0;
      sacc[1][i] = p1;
    }

    // ---- P (C-layout) -> LDS -> A-layout fragment (wave-private) ----
    unsigned short (*P)[32] = Ps[wave];
#pragma unroll
    for (int kt = 0; kt < 2; ++kt)
#pragma unroll
      for (int i = 0; i < 8; ++i)
        P[hg * 8 + i][kt * 16 + lr] = f32_to_bf16(sacc[kt][i]);

    const v16bf pf = ld_frag(&P[lr][hg * 8], &P[lr][16 + hg * 8]);

    // ---- O += P @ V  (4 d-tiles; V^T staged so B-fragment is contiguous) ----
#pragma unroll
    for (int dt = 0; dt < 4; ++dt) {
      const v16bf vf = ld_frag(&Vts[cur][dt * 16 + lr][hg * 8],
                               &Vts[cur][dt * 16 + lr][16 + hg * 8]);
      oacc[dt] = wmma_bf16(pf, vf, oacc[dt]);
    }
  }

  // ---- finalize: divide by row sum, write bf16 as [b,s,(h,d)] ----
#pragma unroll
  for (int i = 0; i < 8; ++i) {
    float l = lrow[i];
#pragma unroll
    for (int moff = 8; moff >= 1; moff >>= 1)
      l += __shfl_xor(l, moff, 32);
    const float inv = 1.0f / l;
    const int q = q0 + wave * 16 + hg * 8 + i;
    const size_t base = ((size_t)b * SEQ + q) * EMB + h * HD;
#pragma unroll
    for (int dt = 0; dt < 4; ++dt)
      attn_ws[base + dt * 16 + lr] = f32_to_bf16(oacc[dt][i] * inv);
  }
}

// ---------------------------------------------------------------------------
// Kernel 3: output projection.  out = attn @ Wo.T, fp32 result.
// grid = (MTOT/128, EMB/64)
// ---------------------------------------------------------------------------
__global__ __launch_bounds__(256)
void o_proj_kernel(const unsigned short* __restrict__ attn,
                   const float* __restrict__ Wo,
                   float* __restrict__ out) {
  __shared__ unsigned short xs[128][32];
  __shared__ unsigned short wsh[64][32];

  const int m0 = blockIdx.x * 128;
  const int n0 = blockIdx.y * 64;

  const int tid  = threadIdx.x;
  const int wave = tid >> 5;
  const int lane = tid & 31;
  const int lr   = lane & 15;
  const int hg   = (lane >> 4) & 1;

  v8f acc[4];
#pragma unroll
  for (int i = 0; i < 4; ++i) acc[i] = (v8f)0.0f;

  for (int k0 = 0; k0 < EMB; k0 += 32) {
    // ---- stage activation tile (already bf16 -> async DMA to LDS) ----
    {
      const int r  = tid >> 1;
      const int c0 = (tid & 1) * 16;
      const unsigned short* g = attn + (size_t)(m0 + r) * EMB + k0 + c0;
      unsigned short* l = &xs[r][c0];
#if ASYNC_LDS
      ASYNC_CP16(g, l, 0);
      ASYNC_CP16(g, l, 16);
#else
      const uint4* src = (const uint4*)g;
      *(uint4*)&xs[r][c0]     = src[0];
      *(uint4*)&xs[r][c0 + 8] = src[1];
#endif
    }
    // ---- stage W tile: f32 -> bf16 (packed cvt) ----
    {
      const int r  = tid >> 2;
      const int c0 = (tid & 3) * 8;
      const float* src = Wo + (size_t)(n0 + r) * EMB + k0 + c0;
      alignas(16) unsigned int t32[4];
#pragma unroll
      for (int j = 0; j < 2; ++j) {
        float4 f = *(const float4*)(src + j * 4);
        t32[j * 2 + 0] = pack2_bf16(f.x, f.y);
        t32[j * 2 + 1] = pack2_bf16(f.z, f.w);
      }
      *(uint4*)&wsh[r][c0] = *(const uint4*)&t32[0];
    }
    WAIT_ASYNC();
    __syncthreads();

    const v16bf a = ld_frag(&xs[wave * 16 + lr][hg * 8],
                            &xs[wave * 16 + lr][16 + hg * 8]);
#pragma unroll
    for (int nt = 0; nt < 4; ++nt) {
      const v16bf b = ld_frag(&wsh[nt * 16 + lr][hg * 8],
                              &wsh[nt * 16 + lr][16 + hg * 8]);
      acc[nt] = wmma_bf16(a, b, acc[nt]);
    }
    __syncthreads();
  }

  {
    float* obase = out + (size_t)(m0 + wave * 16 + hg * 8) * EMB + n0 + lr;
#pragma unroll
    for (int nt = 0; nt < 4; ++nt)
#pragma unroll
      for (int i = 0; i < 8; ++i)
        obase[(size_t)i * EMB + nt * 16] = acc[nt][i];
  }
}

// ---------------------------------------------------------------------------
extern "C" void kernel_launch(void* const* d_in, const int* in_sizes, int n_in,
                              void* d_out, int out_size, void* d_ws,
                              size_t ws_size, hipStream_t stream) {
  (void)in_sizes; (void)n_in; (void)out_size; (void)ws_size;

  const float* x    = (const float*)d_in[0];
  const float* bias = (const float*)d_in[1];
  const float* Wq   = (const float*)d_in[2];
  const float* Wk   = (const float*)d_in[3];
  const float* Wv   = (const float*)d_in[4];
  const float* Wo   = (const float*)d_in[5];
  float* out = (float*)d_out;

  const size_t HSD = (size_t)BSZ * NH * SEQ * HD;   // 8,388,608 elems
  unsigned short* q_ws    = (unsigned short*)d_ws;
  unsigned short* k_ws    = q_ws + HSD;
  unsigned short* v_ws    = k_ws + HSD;
  unsigned short* attn_ws = v_ws + HSD;

  dim3 g1(MTOT / 128, EMB / 64, 3);
  qkv_proj_kernel<<<g1, 256, 0, stream>>>(x, Wq, Wk, Wv, q_ws, k_ws, v_ws);

  dim3 g2(SEQ / 128, NH, BSZ);
  attn_kernel<<<g2, 256, 0, stream>>>(q_ws, k_ws, v_ws, bias, attn_ws);

  dim3 g3(MTOT / 128, EMB / 64, 1);
  o_proj_kernel<<<g3, 256, 0, stream>>>(attn_ws, Wo, out);
}
